// MultiHeadedAttention_58085137711358
// MI455X (gfx1250) — matrix operand order repaired and checked by reading, not compiled, not run on hardware
//
#include <hip/hip_runtime.h>
#include <hip/hip_bf16.h>

typedef __attribute__((ext_vector_type(16))) _Float16 v16h;
typedef __attribute__((ext_vector_type(8)))  _Float16 v8h;
typedef __attribute__((ext_vector_type(4)))  _Float16 v4h;
typedef __attribute__((ext_vector_type(8)))  float    v8f;
typedef __attribute__((ext_vector_type(4)))  float    v4f;

#define DEVINL __device__ __forceinline__

static constexpr int S_LEN = 2048;
static constexpr int DM    = 1024;
static constexpr int NH    = 16;
static constexpr int DK    = 64;
static constexpr int NB    = 2;
static constexpr int MROWS = NB * S_LEN; // 4096

// ---- WMMA fragment loaders (LDS, row-major tiles) -------------------------
// A fragment: 16(M) x 32(K) f16.  ISA layout: lane m=L&15, g=L>>4;
// halves 0..7 -> K = 8g..8g+7 ; halves 8..15 -> K = 16+8g..16+8g+7.
DEVINL v16h frag_a(const _Float16* base, int ld) {
  int lane = threadIdx.x & 31;
  int g = lane >> 4, m = lane & 15;
  const _Float16* p = base + m * ld + 8 * g;
  v8h lo = *(const v8h*)(p);        // ds_load_b128
  v8h hi = *(const v8h*)(p + 16);   // ds_load_b128
  v16h r;
#pragma unroll
  for (int i = 0; i < 8; ++i) { r[i] = lo[i]; r[8 + i] = hi[i]; }
  return r;
}

// B fragment: 32(K) x 16(N) f16, sourced from a row-major [n][k] tile
// (i.e. B[k][n] = tile[n][k]).  Lane n=L&15, g=L>>4; halves 0..7 -> K=8g..8g+7 ; halves 8..15 -> K=16+8g..16+8g+7.
DEVINL v16h frag_bt(const _Float16* base, int ld) {
  int lane = threadIdx.x & 31;
  int g = lane >> 4, n = lane & 15;
  const _Float16* p = base + n * ld + 8 * g;
  v8h lo = *(const v8h*)(p);        // ds_load_b128
  v8h hi = *(const v8h*)(p + 16);   // ds_load_b128
  v16h r;
#pragma unroll
  for (int i = 0; i < 8; ++i) { r[i] = lo[i]; r[8 + i] = hi[i]; }
  return r;
}

DEVINL v8f wmma_f16(v16h a, v16h b, v8f c) {
  return __builtin_amdgcn_wmma_f32_16x16x32_f16(false, a, false, b, (short)0, c,
                                                false, false);
}

// ---- GEMM:  C[M,N] = A[M,K] @ W[N,K]^T + bias[N] --------------------------
// 128x128 tile / block, BK=32, 8 waves each computing 32x64.
template <bool A_F16, bool OUT_F16>
__global__ __launch_bounds__(256) void gemm_nt_kernel(
    const void* __restrict__ Aptr, const float* __restrict__ W,
    const float* __restrict__ bias, void* __restrict__ Cptr,
    int M, int N, int K) {
  constexpr int LDT = 40; // 32 + 8 pad, keeps 16B alignment (80B rows)
  __shared__ __align__(16) _Float16 sA[128 * LDT];
  __shared__ __align__(16) _Float16 sW[128 * LDT];

  const int tid  = threadIdx.x;
  const int wave = tid >> 5;
  const int m0 = blockIdx.y * 128;
  const int n0 = blockIdx.x * 128;
  const int wm = (wave & 3) * 32;   // wave row offset
  const int wn = (wave >> 2) * 64;  // wave col offset

  v8f zero = {};
  v8f acc[2][4];
#pragma unroll
  for (int i = 0; i < 2; ++i)
#pragma unroll
    for (int j = 0; j < 4; ++j) acc[i][j] = zero;

  const int lr = tid >> 3;       // 0..31
  const int lc = (tid & 7) * 4;  // 0..28 step 4

  for (int k0 = 0; k0 < K; k0 += 32) {
    __syncthreads();
#pragma unroll
    for (int it = 0; it < 4; ++it) {
      int row = it * 32 + lr;
      v4h ha, hw;
      if constexpr (A_F16) {
        ha = *(const v4h*)((const _Float16*)Aptr + (size_t)(m0 + row) * K + k0 + lc);
      } else {
        v4f fa = *(const v4f*)((const float*)Aptr + (size_t)(m0 + row) * K + k0 + lc);
#pragma unroll
        for (int j = 0; j < 4; ++j) ha[j] = (_Float16)fa[j];
      }
      v4f fw = *(const v4f*)(W + (size_t)(n0 + row) * K + k0 + lc);
#pragma unroll
      for (int j = 0; j < 4; ++j) hw[j] = (_Float16)fw[j];
      *(v4h*)(sA + row * LDT + lc) = ha;
      *(v4h*)(sW + row * LDT + lc) = hw;
    }
    __syncthreads();

    v16h af[2], bf[4];
#pragma unroll
    for (int i = 0; i < 2; ++i) af[i] = frag_a(sA + (wm + 16 * i) * LDT, LDT);
#pragma unroll
    for (int j = 0; j < 4; ++j) bf[j] = frag_bt(sW + (wn + 16 * j) * LDT, LDT);
#pragma unroll
    for (int i = 0; i < 2; ++i)
#pragma unroll
      for (int j = 0; j < 4; ++j) acc[i][j] = wmma_f16(af[i], bf[j], acc[i][j]);
  }

  const int lane = tid & 31;
  const int g = lane >> 4, ln = lane & 15;
#pragma unroll
  for (int j = 0; j < 4; ++j) {
    int col = n0 + wn + 16 * j + ln;
    float bv = bias[col];
#pragma unroll
    for (int i = 0; i < 2; ++i) {
#pragma unroll
      for (int r = 0; r < 8; ++r) {
        int row = m0 + wm + 16 * i + r + 8 * g;
        float v = acc[i][j][r] + bv;
        if constexpr (OUT_F16)
          ((_Float16*)Cptr)[(size_t)row * N + col] = (_Float16)v;
        else
          ((float*)Cptr)[(size_t)row * N + col] = v;
      }
    }
  }
}

// ---- Flash attention: per (b,h), 128 Q rows / block, 64-wide K/V blocks ---
__global__ __launch_bounds__(256) void flash_attn_kernel(
    const _Float16* __restrict__ Q, const _Float16* __restrict__ Kp,
    const _Float16* __restrict__ Vp, const int* __restrict__ mask,
    _Float16* __restrict__ ctx) {
  constexpr int LDQ = 72; // 64 + 8 pad (144B rows, 16B aligned)
  __shared__ __align__(16) _Float16 sQ[128 * LDQ];
  __shared__ __align__(16) _Float16 sK[64 * LDQ];
  __shared__ __align__(16) _Float16 sV[64 * LDQ]; // transposed: [d][k]
  __shared__ __align__(16) _Float16 sP[128 * LDQ];

  const int tid = threadIdx.x;
  const int bh = blockIdx.x;
  const int b = bh >> 4, h = bh & 15;
  const int qbase = blockIdx.y * 128;
  const int wave = tid >> 5;
  const int lane = tid & 31;
  const int g = lane >> 4, ln = lane & 15;
  const int qoff = wave * 16;

  // stage Q tile (128 x 64)
  {
    const int colq = (tid & 15) * 4;
    const int row0 = tid >> 4;
#pragma unroll
    for (int it = 0; it < 8; ++it) {
      int row = row0 + it * 16;
      *(v4h*)(sQ + row * LDQ + colq) =
          *(const v4h*)(Q + (size_t)(b * S_LEN + qbase + row) * DM + h * DK + colq);
    }
  }

  float mrow[8], lrow[8];
#pragma unroll
  for (int r = 0; r < 8; ++r) { mrow[r] = -1e30f; lrow[r] = 0.0f; }
  v8f zero = {};
  v8f o[4];
#pragma unroll
  for (int d = 0; d < 4; ++d) o[d] = zero;

  __syncthreads();
  const v16h qa0 = frag_a(sQ + qoff * LDQ + 0, LDQ);
  const v16h qa1 = frag_a(sQ + qoff * LDQ + 32, LDQ);

  const int* mbase = mask + (size_t)b * S_LEN * S_LEN;

  for (int kb = 0; kb < S_LEN; kb += 64) {
    __syncthreads();
    { // stage K (row-major) and V (transposed) 64x64 tiles
      const int colq = (tid & 15) * 4;
      const int row0 = tid >> 4;
#pragma unroll
      for (int it = 0; it < 4; ++it) {
        int row = row0 + it * 16;
        *(v4h*)(sK + row * LDQ + colq) =
            *(const v4h*)(Kp + (size_t)(b * S_LEN + kb + row) * DM + h * DK + colq);
        v4h vv = *(const v4h*)(Vp + (size_t)(b * S_LEN + kb + row) * DM + h * DK + colq);
#pragma unroll
        for (int j = 0; j < 4; ++j) sV[(colq + j) * LDQ + row] = vv[j];
      }
    }
    __syncthreads();

    // scores: 16(q) x 64(k) per wave
    v8f st[4];
#pragma unroll
    for (int nt = 0; nt < 4; ++nt) {
      st[nt] = zero;
      st[nt] = wmma_f16(qa0, frag_bt(sK + nt * 16 * LDQ + 0, LDQ), st[nt]);
      st[nt] = wmma_f16(qa1, frag_bt(sK + nt * 16 * LDQ + 32, LDQ), st[nt]);
    }

    // mask then scale by 1/sqrt(64) (matches reference order: -1e9 then /8)
#pragma unroll
    for (int nt = 0; nt < 4; ++nt) {
      int kg = kb + nt * 16 + ln;
#pragma unroll
      for (int r = 0; r < 8; ++r) {
        int qg = qbase + qoff + r + 8 * g;
        int mv = mbase[(size_t)qg * S_LEN + kg];
        float s = st[nt][r];
        st[nt][r] = mv ? s * 0.125f : -1.25e8f;
      }
    }

    // online softmax: each 16-lane half holds one full row per acc VGPR
#pragma unroll
    for (int r = 0; r < 8; ++r) {
      float v = fmaxf(fmaxf(st[0][r], st[1][r]), fmaxf(st[2][r], st[3][r]));
#pragma unroll
      for (int off = 8; off >= 1; off >>= 1) v = fmaxf(v, __shfl_xor(v, off, 16));
      float mn = fmaxf(mrow[r], v);
      float alpha = __expf(mrow[r] - mn);
      mrow[r] = mn;
      float ssum = 0.0f;
#pragma unroll
      for (int nt = 0; nt < 4; ++nt) {
        float p = __expf(st[nt][r] - mn);
        st[nt][r] = p;
        ssum += p;
      }
#pragma unroll
      for (int off = 8; off >= 1; off >>= 1) ssum += __shfl_xor(ssum, off, 16);
      lrow[r] = lrow[r] * alpha + ssum;
#pragma unroll
      for (int d = 0; d < 4; ++d) o[d][r] *= alpha;
    }

    // P (C-layout) -> wave-private LDS -> A-fragment layout
#pragma unroll
    for (int nt = 0; nt < 4; ++nt)
#pragma unroll
      for (int r = 0; r < 8; ++r)
        sP[(qoff + r + 8 * g) * LDQ + nt * 16 + ln] = (_Float16)st[nt][r];
    asm volatile("s_wait_dscnt 0x0" : : : "memory"); // wave-local DS RAW fence

    v16h pa0 = frag_a(sP + qoff * LDQ + 0, LDQ);
    v16h pa1 = frag_a(sP + qoff * LDQ + 32, LDQ);
#pragma unroll
    for (int d = 0; d < 4; ++d) {
      o[d] = wmma_f16(pa0, frag_bt(sV + d * 16 * LDQ + 0, LDQ), o[d]);
      o[d] = wmma_f16(pa1, frag_bt(sV + d * 16 * LDQ + 32, LDQ), o[d]);
    }
  }

  // epilogue: normalize and write ctx as f16 [b, q, h*64+d]
#pragma unroll
  for (int r = 0; r < 8; ++r) {
    float inv = 1.0f / fmaxf(lrow[r], 1e-20f);
    int row = qbase + qoff + r + 8 * g;
#pragma unroll
    for (int d = 0; d < 4; ++d) {
      int col = h * DK + d * 16 + ln;
      ctx[(size_t)(b * S_LEN + row) * DM + col] = (_Float16)(o[d][r] * inv);
    }
  }
}

// ---- host side ------------------------------------------------------------
extern "C" void kernel_launch(void* const* d_in, const int* in_sizes, int n_in,
                              void* d_out, int out_size, void* d_ws, size_t ws_size,
                              hipStream_t stream) {
  (void)in_sizes; (void)n_in; (void)out_size; (void)ws_size;
  const float* query = (const float*)d_in[0];
  const float* key   = (const float*)d_in[1];
  const float* value = (const float*)d_in[2];
  const int*   mask  = (const int*)d_in[3];
  const float* w_q = (const float*)d_in[4];
  const float* b_q = (const float*)d_in[5];
  const float* w_k = (const float*)d_in[6];
  const float* b_k = (const float*)d_in[7];
  const float* w_v = (const float*)d_in[8];
  const float* b_v = (const float*)d_in[9];
  const float* w_o = (const float*)d_in[10];
  const float* b_o = (const float*)d_in[11];

  const size_t elems = (size_t)MROWS * DM;
  _Float16* Qh = (_Float16*)d_ws;
  _Float16* Kh = Qh + elems;
  _Float16* Vh = Kh + elems;
  _Float16* Ch = Vh + elems; // total 32 MB of workspace

  dim3 ggrid(DM / 128, MROWS / 128); // (8, 32)
  gemm_nt_kernel<false, true><<<ggrid, 256, 0, stream>>>(query, w_q, b_q, Qh, MROWS, DM, DM);
  gemm_nt_kernel<false, true><<<ggrid, 256, 0, stream>>>(key,   w_k, b_k, Kh, MROWS, DM, DM);
  gemm_nt_kernel<false, true><<<ggrid, 256, 0, stream>>>(value, w_v, b_v, Vh, MROWS, DM, DM);

  dim3 fgrid(NB * NH, S_LEN / 128); // (32, 16)
  flash_attn_kernel<<<fgrid, 256, 0, stream>>>(Qh, Kh, Vh, mask, Ch);

  gemm_nt_kernel<true, false><<<ggrid, 256, 0, stream>>>(Ch, w_o, b_o, (float*)d_out, MROWS, DM, DM);
}
